// DVRJANET_48988396978774
// MI455X (gfx1250) — matrix operand order repaired and checked by reading, not compiled, not run on hardware
//
#include <hip/hip_runtime.h>

#define BB 64
#define TT 1024
#define HH 256
#define NWG 16
#define NTHREADS 512   // 16 waves of 32

typedef __attribute__((ext_vector_type(16))) __bf16 v16bf;
typedef __attribute__((ext_vector_type(8)))  float  v8f;

// LDS element counts
#define SW1_ELEMS (3*8*32*16)    // Wah/Wph/Wf B-fragments: [mat][kt][lane][16] bf16
#define SW2_ELEMS (2*16*32*16)   // Wgc/Wgs   B-fragments: [mat][kt][lane][16] bf16
#define SP1_ELEMS (3*64*16)      // phase-1 accumulators [mat][b][n] f32
#define SF_ELEMS  (64*16)        // forget gate slice f32
#define SMEM_BYTES (size_t)((SW1_ELEMS + SW2_ELEMS)*2 + (SP1_ELEMS + SF_ELEMS)*4)

__device__ __forceinline__ float sigmoidf_(float v) { return 1.0f / (1.0f + __expf(-v)); }

// Device-wide phased barrier: counter in L2, agent-scope polling + s_sleep.
__device__ __forceinline__ void gbar(unsigned* bar, unsigned& tgt) {
    __syncthreads();
    if (threadIdx.x == 0) {
        __threadfence();                 // publish prior global writes
        tgt += NWG;
        atomicAdd(bar, 1u);
        while (__hip_atomic_load(bar, __ATOMIC_RELAXED, __HIP_MEMORY_SCOPE_AGENT) < tgt) {
            __builtin_amdgcn_s_sleep(1);
        }
        __threadfence();                 // acquire others' writes
    }
    __syncthreads();
}

__global__ void __launch_bounds__(NTHREADS)
dvrjanet_scan(const float* __restrict__ x,     // [B,T,2]
              const float* __restrict__ hI0,   // [B,H]
              const float* __restrict__ hQ0,
              const float* __restrict__ c1p, const float* __restrict__ c2p,
              const float* __restrict__ c3p,
              const float* __restrict__ Wa,  const float* __restrict__ Wah,
              const float* __restrict__ Wp1, const float* __restrict__ Wph,
              const float* __restrict__ Wf,  const float* __restrict__ bfv,
              const float* __restrict__ Wgc, const float* __restrict__ bgc,
              const float* __restrict__ Wgs, const float* __restrict__ bgs,
              const float* __restrict__ WI,  const float* __restrict__ bIp,
              const float* __restrict__ WQ,  const float* __restrict__ bQp,
              float* __restrict__ out,          // [B,T,2], pre-zeroed
              unsigned* __restrict__ bar,       // barrier counter (zeroed)
              float* __restrict__ hbuf,         // [parity2][IQ2][B*H] f32
              unsigned short* __restrict__ gcw, // [B*H] bf16 (cos(theta)*anew)
              unsigned short* __restrict__ gsw) // [B*H] bf16 (sin(theta)*anew)
{
    extern __shared__ __align__(32) char smem[];
    __bf16* sW1 = (__bf16*)smem;
    __bf16* sW2 = sW1 + SW1_ELEMS;
    float*  sP1 = (float*)(sW2 + SW2_ELEMS);
    float*  sF  = sP1 + SP1_ELEMS;

    __bf16* gcosb = (__bf16*)gcw;
    __bf16* gsinb = (__bf16*)gsw;

    const int g    = blockIdx.x;          // this WG owns N columns [16g, 16g+16)
    const int tid  = threadIdx.x;
    const int wave = tid >> 5;
    const int lane = tid & 31;
    const int r    = lane & 15;           // A-fragment row / C-fragment N
    const int h    = lane >> 4;           // half-wave selector
    unsigned  tgt  = 0;

    const float c1 = c1p[0], c2 = c2p[0], c3 = c3p[0];
    const float csum = c1 + c2 + c3;
    const float coff = c1 * (1.0f/3.0f) + c2 * (2.0f/3.0f) + c3;
    const float bI = bIp[0], bQ = bQp[0];

    // ---- one-time: scatter this WG's weight column-slices into LDS as
    //      per-lane-contiguous bf16 WMMA B-fragments ----
    for (int idx = tid; idx < SW1_ELEMS; idx += NTHREADS) {
        int j = idx & 15, l = (idx >> 4) & 31, kt = (idx >> 9) & 7, mat = idx >> 12;
        int k = 32*kt + 8*(l >> 4) + (j & 7) + 16*(j >> 3);  // B frag: K = {8h+j} U {16+8h+j}
        int n = 16*g + (l & 15);
        const float* W = (mat == 0) ? Wah : (mat == 1) ? Wph : Wf;
        sW1[idx] = (__bf16)W[k*HH + n];
    }
    for (int idx = tid; idx < SW2_ELEMS; idx += NTHREADS) {
        int j = idx & 15, l = (idx >> 4) & 31, kt = (idx >> 9) & 15, mat = idx >> 13;
        int k = 32*kt + 8*(l >> 4) + (j & 7) + 16*(j >> 3);  // K in [0,512)
        int n = 16*g + (l & 15);
        const float* W = mat ? Wgs : Wgc;
        sW2[idx] = (__bf16)W[k*HH + n];
    }
    // ---- init recurrent state slice (parity 0) ----
    for (int e = tid; e < BB*16; e += NTHREADS) {
        int b = e >> 4, n = 16*g + (e & 15);
        hbuf[b*HH + n]           = hI0[b*HH + n];
        hbuf[BB*HH + b*HH + n]   = hQ0[b*HH + n];
    }
    gbar(bar, tgt);

    for (int t = 0; t < TT; ++t) {
        const float* hIin = hbuf + (t & 1) * 2*BB*HH;
        const float* hQin = hIin + BB*HH;
        float* hIout = hbuf + ((t & 1) ^ 1) * 2*BB*HH;
        float* hQout = hIout + BB*HH;

        // ===== phase 1: a/theta/f pre-activations; 12 tile-jobs over waves =====
        if (wave < 12) {
            const int mat = wave >> 2, mt = wave & 3;   // mat: 0=Wah 1=Wph 2=Wf
            v8f acc = {0.f,0.f,0.f,0.f,0.f,0.f,0.f,0.f};
            const float* pI = hIin + (16*mt + r)*HH + 8*h;
            const float* pQ = hQin + (16*mt + r)*HH + 8*h;
            #pragma unroll
            for (int kt = 0; kt < 8; ++kt) {
                v16bf a;
                #pragma unroll
                for (int j = 0; j < 8; ++j) {   // A lane layout: K = {8h+j} U {16+8h+j}
                    a[j]     = (__bf16)(pI[32*kt + j]      * pQ[32*kt + j]);
                    a[j + 8] = (__bf16)(pI[32*kt + 16 + j] * pQ[32*kt + 16 + j]);
                }
                v16bf bm = *(const v16bf*)(sW1 + ((mat*8 + kt)*32 + lane)*16);
                acc = __builtin_amdgcn_wmma_f32_16x16x32_bf16(
                          false, a, false, bm, (short)0, acc, false, false);
            }
            #pragma unroll
            for (int v = 0; v < 8; ++v) {       // C layout: M = v + 8h, N = lane&15
                int b = 16*mt + v + 8*h;
                sP1[(mat*BB + b)*16 + r] = acc[v];
            }
        }
        __syncthreads();

        // ===== elementwise: envelope, phase, forget gate; publish bf16 cos/sin =====
        for (int e = tid; e < BB*16; e += NTHREADS) {
            int b = e >> 4, n = e & 15, ng = 16*g + n;
            float x0 = x[(b*TT + t)*2 + 0];
            float x1 = x[(b*TT + t)*2 + 1];
            float a    = x0 * Wa[ng]  + sP1[(0*BB + b)*16 + n];
            float anew = a * csum - coff;
            float th   = x1 * Wp1[ng] + sP1[(1*BB + b)*16 + n];
            float fv   = sigmoidf_(sP1[(2*BB + b)*16 + n] + bfv[ng]);
            sF[e] = fv;
            gcosb[b*HH + ng] = (__bf16)(__cosf(th) * anew);
            gsinb[b*HH + ng] = (__bf16)(__sinf(th) * anew);
            if (t + 1 < TT) __builtin_prefetch(&x[(b*TT + t + 1)*2], 0, 1);
        }
        gbar(bar, tgt);

        // ===== phase 2: gated candidates, state update, fused output proj =====
        if (wave < 8) {
            const int mat = wave >> 2, mt = wave & 3;   // mat 0 -> hI/I, 1 -> hQ/Q
            const int ng = 16*g + r;
            const float*  hin  = mat ? hQin  : hIin;
            const __bf16* gb   = mat ? gsinb : gcosb;
            float bias = mat ? bgs[ng] : bgc[ng];
            v8f acc = {bias,bias,bias,bias,bias,bias,bias,bias};
            const float*  hrow = hin + (16*mt + r)*HH + 8*h;
            const __bf16* grow = gb  + (16*mt + r)*HH + 8*h;
            #pragma unroll
            for (int kt = 0; kt < 16; ++kt) {           // K = [h | cos*a] = 512
                v16bf a;
                if (kt < 8) {
                    #pragma unroll
                    for (int j = 0; j < 8; ++j) {
                        a[j]     = (__bf16)hrow[32*kt + j];
                        a[j + 8] = (__bf16)hrow[32*kt + 16 + j];
                    }
                } else {
                    #pragma unroll
                    for (int j = 0; j < 8; ++j) {
                        a[j]     = grow[32*(kt - 8) + j];
                        a[j + 8] = grow[32*(kt - 8) + 16 + j];
                    }
                }
                v16bf bm = *(const v16bf*)(sW2 + ((mat*16 + kt)*32 + lane)*16);
                acc = __builtin_amdgcn_wmma_f32_16x16x32_bf16(
                          false, a, false, bm, (short)0, acc, false, false);
            }
            float  wv   = mat ? WQ[ng] : WI[ng];
            float* hout = mat ? hQout  : hIout;
            #pragma unroll
            for (int v = 0; v < 8; ++v) {
                int b = 16*mt + v + 8*h;
                float gval = tanhf(acc[v]);
                float fv   = sF[b*16 + r];
                float hold = hin[b*HH + ng];
                float hnew = fv * hold + (1.0f - fv) * gval;
                hout[b*HH + ng] = hnew;
                // fused projection: reduce over the 16 N-lanes of this group
                float s = hnew * wv;
                s += __shfl_xor(s, 1, 32);
                s += __shfl_xor(s, 2, 32);
                s += __shfl_xor(s, 4, 32);
                s += __shfl_xor(s, 8, 32);
                if ((lane & 15) == 0) {
                    float add = s + ((g == 0) ? (mat ? bQ : bI) : 0.0f);
                    atomicAdd(&out[(b*TT + t)*2 + mat], add);
                }
            }
        }
        gbar(bar, tgt);
    }
}

extern "C" void kernel_launch(void* const* d_in, const int* in_sizes, int n_in,
                              void* d_out, int out_size, void* d_ws, size_t ws_size,
                              hipStream_t stream) {
    (void)in_sizes; (void)n_in; (void)ws_size;
    const float* x   = (const float*)d_in[0];
    const float* hI0 = (const float*)d_in[1];
    const float* hQ0 = (const float*)d_in[2];
    const float* c1  = (const float*)d_in[3];
    const float* c2  = (const float*)d_in[4];
    const float* c3  = (const float*)d_in[5];
    const float* Wa  = (const float*)d_in[6];
    const float* Wah = (const float*)d_in[7];
    const float* Wp1 = (const float*)d_in[8];
    const float* Wph = (const float*)d_in[9];
    const float* Wf  = (const float*)d_in[10];
    const float* bf_ = (const float*)d_in[11];
    const float* Wgc = (const float*)d_in[12];
    const float* bgc = (const float*)d_in[13];
    const float* Wgs = (const float*)d_in[14];
    const float* bgs = (const float*)d_in[15];
    const float* WI  = (const float*)d_in[16];
    const float* bI  = (const float*)d_in[17];
    const float* WQ  = (const float*)d_in[18];
    const float* bQ  = (const float*)d_in[19];

    char* ws = (char*)d_ws;
    unsigned* bar = (unsigned*)ws;                               // 256 B header
    float* hbuf = (float*)(ws + 256);                            // 2*2*B*H f32
    unsigned short* gcw = (unsigned short*)(ws + 256 + (size_t)4*BB*HH*sizeof(float));
    unsigned short* gsw = gcw + (size_t)BB*HH;

    hipMemsetAsync(bar, 0, 256, stream);
    hipMemsetAsync(d_out, 0, (size_t)out_size * sizeof(float), stream);

    dvrjanet_scan<<<NWG, NTHREADS, SMEM_BYTES, stream>>>(
        x, hI0, hQ0, c1, c2, c3, Wa, Wah, Wp1, Wph, Wf, bf_,
        Wgc, bgc, Wgs, bgs, WI, bI, WQ, bQ,
        (float*)d_out, bar, hbuf, gcw, gsw);
}
